// OptimizedCPMLoss_5746666242354
// MI455X (gfx1250) — matrix operand order repaired and checked by reading, not compiled, not run on hardware
//
#include <hip/hip_runtime.h>
#include <hip/hip_bf16.h>

#define N_BRANCHES  4
#define BATCH       8192
#define DIM         4096
#define NUM_CLASSES 1024
#define MARGIN      0.3f
#define LOSS_EPS    1e-8f
#define N_PAIRS     6

typedef __bf16 v16bf __attribute__((ext_vector_type(16)));
typedef float  v8f   __attribute__((ext_vector_type(8)));

union BF16Tile { uint4 q[2]; v16bf v; };

__device__ __forceinline__ unsigned short f32_to_bf16_rne(float f) {
    unsigned u = __float_as_uint(f);
    unsigned r = u + 0x7FFFu + ((u >> 16) & 1u);
    return (unsigned short)(r >> 16);
}

// CDNA5 async global->LDS copy (ISA 15.18.3 op 98). VDST = LDS byte offset
// (generic pointers to __shared__ carry the group-relative offset in the low
// 32 bits), VADDR = 64-bit global address. Tracked by ASYNCcnt.
__device__ __forceinline__ void async_copy_b128(void* lds_dst, const void* gsrc) {
    unsigned ldsoff = (unsigned)(uintptr_t)lds_dst;
    unsigned long long ga = (unsigned long long)(uintptr_t)gsrc;
    asm volatile("global_load_async_to_lds_b128 %0, %1, off"
                 :: "v"(ldsoff), "v"(ga) : "memory");
}
__device__ __forceinline__ void wait_async0() {
    asm volatile("s_wait_asynccnt 0x0" ::: "memory");
}

// ---------------------------------------------------------------- init
__global__ void k_init(float* counts, unsigned* hn, float* out, int out_size) {
    int t = blockIdx.x * blockDim.x + threadIdx.x;
    int stride = gridDim.x * blockDim.x;
    for (int i = t; i < NUM_CLASSES; i += stride) counts[i] = 0.0f;
    for (int i = t; i < N_BRANCHES * NUM_CLASSES; i += stride) hn[i] = 0x7F800000u; // +inf
    for (int i = t; i < out_size; i += stride) out[i] = 0.0f;
}

// ---------------------------------------------------------------- counts
__global__ void k_counts(const int* __restrict__ targets, float* counts) {
    int i = blockIdx.x * blockDim.x + threadIdx.x;
    if (i < BATCH) atomicAdd(&counts[targets[i]], 1.0f);
}

// ---------------------------------------------------------------- centers (segment mean)
// grid = (DIM/16, N_BRANCHES), block = 256.  LDS: 1024 classes x 16 dims = 64KB.
// This is the single pass over the 512MB input -> HBM-bound (~22us floor).
__global__ void __launch_bounds__(256) k_centers(const float* __restrict__ feats,
                                                 const int* __restrict__ targets,
                                                 const float* __restrict__ counts,
                                                 float* __restrict__ cf32,
                                                 unsigned short* __restrict__ cbf) {
    __shared__ float acc[NUM_CLASSES * 16];
    const int chunk = blockIdx.x;      // dims [chunk*16, chunk*16+16)
    const int b     = blockIdx.y;
    const int tid   = threadIdx.x;
    const int dl    = tid & 15;
    const int grp   = tid >> 4;        // 16 row groups

    for (int i = tid; i < NUM_CLASSES * 16; i += 256) acc[i] = 0.0f;
    __syncthreads();

    const size_t fbase = (size_t)b * BATCH * DIM + (size_t)chunk * 16 + dl;
    for (int r = grp; r < BATCH; r += 16) {
        int c = targets[r];
        float v = feats[fbase + (size_t)r * DIM];
        if (r + 64 < BATCH) __builtin_prefetch(&feats[fbase + (size_t)(r + 64) * DIM], 0, 1);
        atomicAdd(&acc[c * 16 + dl], v);           // ds_add_f32
    }
    __syncthreads();

    for (int i = tid; i < NUM_CLASSES * 16; i += 256) {
        int c = i >> 4, d = i & 15;
        float v = acc[i] / counts[c];
        size_t o = ((size_t)b * NUM_CLASSES + c) * DIM + (size_t)chunk * 16 + d;
        cf32[o] = v;
        cbf[o]  = f32_to_bf16_rne(v);
    }
}

// ---------------------------------------------------------------- squared norms
__global__ void __launch_bounds__(256) k_sqnorm(const float* __restrict__ cf32,
                                                float* __restrict__ sq) {
    __shared__ float red[256];
    const int row = blockIdx.x;                    // b*1024 + c
    const float* p = cf32 + (size_t)row * DIM;
    float s = 0.0f;
    for (int d = threadIdx.x; d < DIM; d += 256) { float v = p[d]; s += v * v; }
    red[threadIdx.x] = s;
    __syncthreads();
    for (int w = 128; w > 0; w >>= 1) {
        if (threadIdx.x < w) red[threadIdx.x] += red[threadIdx.x + w];
        __syncthreads();
    }
    if (threadIdx.x == 0) sq[row] = red[0];
}

// ---------------------------------------------------------------- Gram (WMMA bf16) + hard-neg
// grid = (1024/128, 1024/128, N_BRANCHES), block = 256 (8 waves).
// Block tile 128x128; wave w -> rows [by*128 + w*16, +16), cols [bx*128, +128).
// B block staged in LDS via async copies (double buffered); A direct from L2.
#define KSTEP     64
#define BROW_PAD  144                          // 128B data + 16B pad (16B aligned)
__global__ void __launch_bounds__(256) k_gram_hardneg(const unsigned short* __restrict__ cbf,
                                                      const float* __restrict__ sq,
                                                      unsigned* __restrict__ hn) {
    __shared__ __align__(16) unsigned char Bs[2][128 * BROW_PAD];   // 36 KB

    const int b    = blockIdx.z;
    const int tid  = threadIdx.x;
    const int wave = tid >> 5;
    const int lane = tid & 31;
    const int half = lane >> 4;
    const int lrow = lane & 15;
    const int m0   = blockIdx.y * 128 + wave * 16;
    const int n0   = blockIdx.x * 128;

    const unsigned short* cb    = cbf + (size_t)b * NUM_CLASSES * DIM;
    const unsigned short* Abase = cb + (size_t)(m0 + lrow) * DIM;

    v8f acc[8];
#pragma unroll
    for (int j = 0; j < 8; ++j)
#pragma unroll
        for (int e = 0; e < 8; ++e) acc[j][e] = 0.0f;

    // issue one K-stage of B (128 rows x 64 bf16 = 16KB) as 4 async b128/thread
    auto issue_stage = [&](int buf, int k0) {
#pragma unroll
        for (int t = 0; t < 4; ++t) {
            int chunk = t * 256 + tid;             // 0..1023 chunks of 16B
            int row = chunk >> 3, w = chunk & 7;
            const void* g = cb + (size_t)(n0 + row) * DIM + k0 + w * 8;
            void* l = &Bs[buf][row * BROW_PAD + w * 16];
            async_copy_b128(l, g);
        }
    };

    issue_stage(0, 0);

    const int NSTAGE = DIM / KSTEP;                // 64
    for (int s = 0; s < NSTAGE; ++s) {
        const int k0  = s * KSTEP;
        const int buf = s & 1;
        wait_async0();                 // this wave's part of stage s landed
        __syncthreads();               // stage s complete; buf^1 readers done
        if (s + 1 < NSTAGE) issue_stage(buf ^ 1, k0 + KSTEP);  // overlaps compute

#pragma unroll
        for (int sub = 0; sub < 2; ++sub) {        // two K=32 WMMA sub-steps
            // A 16x32 fragment: lanes 0-15 K 0-15,
            // lanes 16-31 K 16-31 of this k32 chunk (matches B).
            BF16Tile A;
            const unsigned short* Ak = Abase + k0 + sub * 32;
            A.q[0] = *(const uint4*)(Ak + half * 16);
            A.q[1] = *(const uint4*)(Ak + half * 16 + 8);
#pragma unroll
            for (int j = 0; j < 8; ++j) {
                // B 32x16 fragment: lane = column j*16+lrow, half selects K 0-15/16-31
                const uint4* p = (const uint4*)(&Bs[buf][(j * 16 + lrow) * BROW_PAD
                                                         + sub * 64 + half * 32]);
                BF16Tile B;
                B.q[0] = p[0];
                B.q[1] = p[1];
                acc[j] = __builtin_amdgcn_wmma_f32_16x16x32_bf16(
                    false, A.v, false, B.v, (short)0, acc[j], false, false);
            }
        }
    }

    const float INF = __uint_as_float(0x7F800000u);
#pragma unroll
    for (int r = 0; r < 8; ++r) {
        // C/D layout: VGPR r, lanes 0-15 -> M=m0+r, N=lane; lanes 16-31 -> M=m0+8+r
        const int m     = m0 + r + half * 8;
        const float sqm = sq[b * NUM_CLASSES + m];
        float dmin = INF;
#pragma unroll
        for (int j = 0; j < 8; ++j) {
            int n = n0 + j * 16 + lrow;
            float g  = acc[j][r];
            float d2 = sqm + sq[b * NUM_CLASSES + n] - 2.0f * g;
            float d  = sqrtf(fmaxf(d2, 0.0f));
            if (m == n) d = INF;                   // mask diagonal
            dmin = fminf(dmin, d);
        }
#pragma unroll
        for (int ss = 1; ss <= 8; ss <<= 1)
            dmin = fminf(dmin, __shfl_xor(dmin, ss, 32));
        if (lrow == 0)                             // lanes 0 and 16 publish
            atomicMin(&hn[b * NUM_CLASSES + m], __float_as_uint(dmin));
    }
}

// ---------------------------------------------------------------- all-pairs positive dist
// One block per class: load 4 branch centers once, form all 6 pair diffs.
__global__ void __launch_bounds__(256) k_pos_loss(const float* __restrict__ cf32,
                                                  const unsigned* __restrict__ hn,
                                                  float* __restrict__ terms) {
    __shared__ float red[N_PAIRS][256];
    const int c = blockIdx.x;
    float s[N_PAIRS];
#pragma unroll
    for (int p = 0; p < N_PAIRS; ++p) s[p] = 0.0f;

    for (int d = threadIdx.x; d < DIM; d += 256) {
        float v0 = cf32[((size_t)0 * NUM_CLASSES + c) * DIM + d];
        float v1 = cf32[((size_t)1 * NUM_CLASSES + c) * DIM + d];
        float v2 = cf32[((size_t)2 * NUM_CLASSES + c) * DIM + d];
        float v3 = cf32[((size_t)3 * NUM_CLASSES + c) * DIM + d];
        float df;
        df = v0 - v1 + LOSS_EPS; s[0] += df * df;
        df = v0 - v2 + LOSS_EPS; s[1] += df * df;
        df = v0 - v3 + LOSS_EPS; s[2] += df * df;
        df = v1 - v2 + LOSS_EPS; s[3] += df * df;
        df = v1 - v3 + LOSS_EPS; s[4] += df * df;
        df = v2 - v3 + LOSS_EPS; s[5] += df * df;
    }
#pragma unroll
    for (int p = 0; p < N_PAIRS; ++p) red[p][threadIdx.x] = s[p];
    __syncthreads();
    for (int w = 128; w > 0; w >>= 1) {
        if (threadIdx.x < w)
#pragma unroll
            for (int p = 0; p < N_PAIRS; ++p) red[p][threadIdx.x] += red[p][threadIdx.x + w];
        __syncthreads();
    }
    if (threadIdx.x == 0) {
        const int PI[N_PAIRS] = {0, 0, 0, 1, 1, 2};
#pragma unroll
        for (int p = 0; p < N_PAIRS; ++p) {
            float pos = sqrtf(red[p][0]);
            float h   = __uint_as_float(hn[PI[p] * NUM_CLASSES + c]);
            float v   = MARGIN + pos - h;
            terms[p * NUM_CLASSES + c] = v > 0.0f ? v : 0.0f;
        }
    }
}

// ---------------------------------------------------------------- deterministic final mean
__global__ void __launch_bounds__(256) k_reduce(const float* __restrict__ terms,
                                                float* __restrict__ out) {
    __shared__ float red[256];
    float s = 0.0f;
    for (int i = threadIdx.x; i < N_PAIRS * NUM_CLASSES; i += 256) s += terms[i];
    red[threadIdx.x] = s;
    __syncthreads();
    for (int w = 128; w > 0; w >>= 1) {
        if (threadIdx.x < w) red[threadIdx.x] += red[threadIdx.x + w];
        __syncthreads();
    }
    if (threadIdx.x == 0) out[0] = red[0] * (1.0f / (float)(N_PAIRS * NUM_CLASSES));
}

// ================================================================ launch
extern "C" void kernel_launch(void* const* d_in, const int* in_sizes, int n_in,
                              void* d_out, int out_size, void* d_ws, size_t ws_size,
                              hipStream_t stream) {
    (void)in_sizes; (void)n_in; (void)ws_size;
    const float* feats   = (const float*)d_in[0];
    const int*   targets = (const int*)d_in[1];
    // d_in[2] = num_classes scalar; baked in as compile-time constant.

    char* ws = (char*)d_ws;
    float*          counts = (float*)ws;                                        //  4 KB
    unsigned*       hn     = (unsigned*)(ws + 4096);                            // 16 KB
    float*          terms  = (float*)(ws + 4096 + 16384);                       // 24 KB
    float*          cf32   = (float*)(ws + (1u << 16));                         // 64 MB
    unsigned short* cbf    = (unsigned short*)(ws + (1u << 16) + (64u << 20));  // 32 MB
    float*          sq     = (float*)(ws + (1u << 16) + (64u << 20) + (32u << 20)); // 16 KB
    float*          out    = (float*)d_out;

    k_init<<<32, 256, 0, stream>>>(counts, hn, out, out_size);
    k_counts<<<BATCH / 256, 256, 0, stream>>>(targets, counts);
    k_centers<<<dim3(DIM / 16, N_BRANCHES), 256, 0, stream>>>(feats, targets, counts, cf32, cbf);
    k_sqnorm<<<N_BRANCHES * NUM_CLASSES, 256, 0, stream>>>(cf32, sq);
    k_gram_hardneg<<<dim3(NUM_CLASSES / 128, NUM_CLASSES / 128, N_BRANCHES), 256, 0, stream>>>(cbf, sq, hn);
    k_pos_loss<<<NUM_CLASSES, 256, 0, stream>>>(cf32, hn, terms);
    k_reduce<<<1, 256, 0, stream>>>(terms, out);
}
